// StructuralLobe_85126251807088
// MI455X (gfx1250) — matrix operand order repaired and checked by reading, not compiled, not run on hardware
//
#include <hip/hip_runtime.h>
#include <hip/hip_bf16.h>

typedef __attribute__((ext_vector_type(16))) __bf16 v16bf;
typedef __attribute__((ext_vector_type(8)))  __bf16 v8bf;
typedef __attribute__((ext_vector_type(2)))  __bf16 v2bf;
typedef __attribute__((ext_vector_type(8)))  float  v8f;

#define DD    640
#define NCAT  1280
#define TT    32768     // 64 * 512 tokens
#define SEQ   512
#define NB    64

#define BROW   80        // padded bytes per LDS tile row (64 data + 16 pad)
#define TILEB  (128*BROW)

// ---------------- CDNA5 async global->LDS copy (guarded) --------------------
#if defined(__gfx1250__) && __has_builtin(__builtin_amdgcn_global_load_async_to_lds_b128)
#define ASYNC_CP 1
#endif

typedef int vi4 __attribute__((vector_size(16)));
typedef __attribute__((address_space(1))) vi4* gas1_t;   // global int4*
typedef __attribute__((address_space(3))) vi4* las3_t;   // LDS int4*

__device__ __forceinline__ void cp_b128(void* lds_dst, const void* gsrc) {
#ifdef ASYNC_CP
    __builtin_amdgcn_global_load_async_to_lds_b128(
        (gas1_t)const_cast<void*>(gsrc), (las3_t)lds_dst, 0, 0);
#else
    *(uint4*)lds_dst = *(const uint4*)gsrc;
#endif
}

__device__ __forceinline__ void cp_wait() {
#ifdef ASYNC_CP
#if __has_builtin(__builtin_amdgcn_s_wait_asynccnt)
    __builtin_amdgcn_s_wait_asynccnt(0);
#else
    asm volatile("s_wait_asynccnt 0" ::: "memory");
#endif
#endif
}

// ---------------------------------------------------------------------------
// K0: convert fp32 weights -> bf16, concatenated [syn ; log] along rows.
// ---------------------------------------------------------------------------
__global__ void k_cvt_weights(const float* __restrict__ syn_in_w,
                              const float* __restrict__ log_in_w,
                              const float* __restrict__ syn_out_w,
                              const float* __restrict__ log_out_w,
                              __bf16* __restrict__ Win,
                              __bf16* __restrict__ Wout) {
    int idx = blockIdx.x * 256 + threadIdx.x;        // 0 .. 2*1280*640-1
    int mat = idx / (NCAT * DD);
    int r   = idx % (NCAT * DD);
    int o   = r / DD;
    int d   = r % DD;
    float v;
    if (mat == 0) v = (o < DD) ? syn_in_w[o * DD + d]  : log_in_w[(o - DD) * DD + d];
    else          v = (o < DD) ? syn_out_w[o * DD + d] : log_out_w[(o - DD) * DD + d];
    (mat == 0 ? Win : Wout)[r] = (__bf16)v;
}

// ---------------------------------------------------------------------------
// K_gates: one wave per token: gate sigmoid + "formal" planes.
// ---------------------------------------------------------------------------
__global__ void k_gates(const float* __restrict__ x,
                        const float* __restrict__ gate_w, const float* __restrict__ gate_b,
                        const float* __restrict__ to_w,   const float* __restrict__ to_b,
                        float* __restrict__ rpl, float* __restrict__ fpl) {
    int gid  = blockIdx.x * blockDim.x + threadIdx.x;
    int t    = gid >> 5;
    int lane = gid & 31;
    const float* xr = x + (size_t)t * DD;
    float sg0 = 0.f, sg1 = 0.f, sg2 = 0.f, sf0 = 0.f, sf1 = 0.f, sf2 = 0.f;
#pragma unroll 4
    for (int i = 0; i < 20; i++) {
        int k = lane + 32 * i;
        float xv = xr[k];
        sg0 = fmaf(xv, gate_w[k],          sg0);
        sg1 = fmaf(xv, gate_w[DD + k],     sg1);
        sg2 = fmaf(xv, gate_w[2 * DD + k], sg2);
        sf0 = fmaf(xv, to_w[k],            sf0);
        sf1 = fmaf(xv, to_w[DD + k],       sf1);
        sf2 = fmaf(xv, to_w[2 * DD + k],   sf2);
    }
#pragma unroll
    for (int off = 16; off; off >>= 1) {
        sg0 += __shfl_xor(sg0, off, 32);
        sg1 += __shfl_xor(sg1, off, 32);
        sg2 += __shfl_xor(sg2, off, 32);
        sf0 += __shfl_xor(sf0, off, 32);
        sf1 += __shfl_xor(sf1, off, 32);
        sf2 += __shfl_xor(sf2, off, 32);
    }
    if (lane == 0) {
        float g[3] = { sg0 + gate_b[0], sg1 + gate_b[1], sg2 + gate_b[2] };
        float s[3] = { sf0 + to_b[0],   sf1 + to_b[1],   sf2 + to_b[2]   };
#pragma unroll
        for (int j = 0; j < 3; j++) {
            rpl[j * TT + t] = 1.f / (1.f + __expf(-g[j]));
            float sym = tanhf(s[j]);
            float la  = __logf(fabsf(sym) + 1e-15f);
            la = fminf(fmaxf(la, -10.f), 10.f);
            fpl[j * TT + t] = __expf(2.f * la);
        }
    }
}

// ---------------------------------------------------------------------------
// K1: u = x (fp32) @ Win^T (bf16)  ->  bf16 u planes.
// 128x128 tile / WG; B tile staged in LDS via async copies (double-buffered);
// B reads software-pipelined one sub-tile ahead of the WMMA that consumes them.
// ---------------------------------------------------------------------------
__global__ void __launch_bounds__(256) k_in_gemm(const float* __restrict__ x,
                                                 const __bf16* __restrict__ Win,
                                                 const float* __restrict__ syn_in_b,
                                                 const float* __restrict__ log_in_b,
                                                 __bf16* __restrict__ u_syn,
                                                 __bf16* __restrict__ u_log) {
    __shared__ alignas(16) char sB[2 * TILEB];            // 20 KB

    const int tid = threadIdx.x;
    const int w = tid >> 5, lane = tid & 31;
    const int half = lane >> 4, lr = lane & 15;
    const int t0 = blockIdx.x * 128 + w * 16;
    const int n0 = blockIdx.y * 128;

    v8f acc[8];
#pragma unroll
    for (int i = 0; i < 8; i++)
#pragma unroll
        for (int e = 0; e < 8; e++) acc[i][e] = 0.f;

    const int arow = t0 + lr;
    const float* xrow = x + (size_t)arow * DD;
    const int koffA = half * 8;
    const int koffB2 = half * 16;                         // byte offset in LDS row

    // stage(kstep, buf): copy 32x128 B tile into LDS (512 x 16B chunks)
    auto stage = [&](int kk, int buf) {
#pragma unroll
        for (int i = 0; i < 2; i++) {
            int c   = tid + 256 * i;                      // 0..511
            int row = c >> 2, seg = c & 3;
            const __bf16* g = Win + (size_t)(n0 + row) * DD + kk + seg * 8;
            cp_b128(sB + buf * TILEB + row * BROW + seg * 16, g);
        }
    };

    stage(0, 0);
    for (int ks = 0; ks < DD / 32; ks++) {
        const int kk = ks * 32;
        cp_wait();
        __syncthreads();
        const int cur = ks & 1;
        if (ks + 1 < DD / 32) stage(kk + 32, cur ^ 1);

        if (kk + 32 < DD) __builtin_prefetch(xrow + kk + 32 + koffA, 0, 3);
        const float* p1 = xrow + kk + koffA;
        const float* p2 = p1 + 16;
        float4 a0 = *(const float4*)p1;
        float4 a1 = *(const float4*)(p1 + 4);
        float4 a2 = *(const float4*)p2;
        float4 a3 = *(const float4*)(p2 + 4);
        v16bf A;
        A[0]  = (__bf16)a0.x; A[1]  = (__bf16)a0.y; A[2]  = (__bf16)a0.z; A[3]  = (__bf16)a0.w;
        A[4]  = (__bf16)a1.x; A[5]  = (__bf16)a1.y; A[6]  = (__bf16)a1.z; A[7]  = (__bf16)a1.w;
        A[8]  = (__bf16)a2.x; A[9]  = (__bf16)a2.y; A[10] = (__bf16)a2.z; A[11] = (__bf16)a2.w;
        A[12] = (__bf16)a3.x; A[13] = (__bf16)a3.y; A[14] = (__bf16)a3.z; A[15] = (__bf16)a3.w;

        auto loadB = [&](int sn) -> v16bf {
            const char* bp = sB + cur * TILEB + (sn * 16 + lr) * BROW + koffB2;
            v8bf b1 = *(const v8bf*)bp;
            v8bf b2 = *(const v8bf*)(bp + 32);
            v16bf B;
#pragma unroll
            for (int i = 0; i < 8; i++) { B[i] = b1[i]; B[8 + i] = b2[i]; }
            return B;
        };

        // software pipeline: fetch B[sn+1] before consuming B[sn]
        v16bf Bcur = loadB(0);
#pragma unroll
        for (int sn = 0; sn < 8; sn++) {
            v16bf Bnext;
            if (sn < 7) Bnext = loadB(sn + 1);
            acc[sn] = __builtin_amdgcn_wmma_f32_16x16x32_bf16(
                false, A, false, Bcur, (short)0, acc[sn], false, false);
            Bcur = Bnext;
        }
    }

    // Epilogue: add bias, convert, scatter to the expert's bf16 plane.
#pragma unroll
    for (int sn = 0; sn < 8; sn++) {
        int n = n0 + sn * 16 + lr;
        float bias = (n < DD) ? syn_in_b[n] : log_in_b[n - DD];
        __bf16* up = (n < DD) ? (u_syn + n) : (u_log + (n - DD));
#pragma unroll
        for (int j = 0; j < 8; j++) {
            int t = t0 + half * 8 + j;
            up[(size_t)t * DD] = (__bf16)(acc[sn][j] + bias);
        }
    }
}

// ---------------------------------------------------------------------------
// K2: decay scan, 8x time-unrolled for memory-level parallelism.
// ---------------------------------------------------------------------------
__global__ void k_scan(const float* __restrict__ syn_ll, const float* __restrict__ log_ll,
                       __bf16* __restrict__ u_syn, __bf16* __restrict__ u_log,
                       float* __restrict__ d_out) {
    int idx = blockIdx.x * 256 + threadIdx.x;       // 2 * 64 * 320
    int e  = idx / (NB * 320);
    int r  = idx % (NB * 320);
    int b  = r / 320;
    int d0 = (r % 320) * 2;
    __bf16* up = e ? u_log : u_syn;
    const float* ll = e ? log_ll : syn_ll;
    float lam0 = __expf(ll[d0]);
    float lam1 = __expf(ll[d0 + 1]);
    float h0 = 0.f, h1 = 0.f;
    __bf16* p = up + (size_t)b * SEQ * DD + d0;
    for (int t = 0; t < SEQ; t += 8) {
        v2bf q[8];
#pragma unroll
        for (int i = 0; i < 8; i++) q[i] = *(const v2bf*)(p + (size_t)i * DD);
#pragma unroll
        for (int i = 0; i < 8; i++) {
            h0 = fmaf(lam0, h0, (float)q[i][0]);
            h1 = fmaf(lam1, h1, (float)q[i][1]);
            q[i][0] = (__bf16)h0;
            q[i][1] = (__bf16)h1;
        }
#pragma unroll
        for (int i = 0; i < 8; i++) *(v2bf*)(p + (size_t)i * DD) = q[i];
        p += (size_t)8 * DD;
    }
    float* st = d_out + (size_t)TT * DD + (size_t)e * NB * DD + (size_t)b * DD + d0;
    st[0] = h0;
    st[1] = h1;
}

// ---------------------------------------------------------------------------
// K3: dual GEMM (syn + log) with LDS-staged, double-buffered B tiles,
// B reads pipelined one sub-tile ahead; fused relevance + formal epilogue.
// ---------------------------------------------------------------------------
__global__ void __launch_bounds__(256) k_out_gemm(const __bf16* __restrict__ h_syn,
                                                  const __bf16* __restrict__ h_log,
                                                  const __bf16* __restrict__ Wout,
                                                  const float* __restrict__ syn_out_b,
                                                  const float* __restrict__ log_out_b,
                                                  const float* __restrict__ ffw,
                                                  const float* __restrict__ ffb,
                                                  const float* __restrict__ rpl,
                                                  const float* __restrict__ fpl,
                                                  float* __restrict__ out) {
    __shared__ alignas(16) char sB[2 * 2 * TILEB];        // 40 KB (2 bufs x 2 experts)

    const int tid = threadIdx.x;
    const int w = tid >> 5, lane = tid & 31;
    const int half = lane >> 4, lr = lane & 15;
    const int t0 = blockIdx.x * 128 + w * 16;
    const int n0 = blockIdx.y * 128;

    v8f accS[8], accL[8];
#pragma unroll
    for (int i = 0; i < 8; i++)
#pragma unroll
        for (int e = 0; e < 8; e++) { accS[i][e] = 0.f; accL[i][e] = 0.f; }

    const int arow = t0 + lr;
    const __bf16* hs = h_syn + (size_t)arow * DD;
    const __bf16* hl = h_log + (size_t)arow * DD;
    const int koffA = half * 8;
    const int koffB2 = half * 16;

    // stage both experts' 32x128 B tiles (1024 x 16B chunks)
    auto stage = [&](int kk, int buf) {
#pragma unroll
        for (int i = 0; i < 4; i++) {
            int c   = tid + 256 * i;                      // 0..1023
            int ex  = c >> 9;
            int rc  = c & 511;
            int row = rc >> 2, seg = rc & 3;
            const __bf16* g = Wout + (size_t)(ex * DD + n0 + row) * DD + kk + seg * 8;
            cp_b128(sB + buf * (2 * TILEB) + ex * TILEB + row * BROW + seg * 16, g);
        }
    };

    stage(0, 0);
    for (int ks = 0; ks < DD / 32; ks++) {
        const int kk = ks * 32;
        cp_wait();
        __syncthreads();
        const int cur = ks & 1;
        if (ks + 1 < DD / 32) stage(kk + 32, cur ^ 1);

        if (kk + 32 < DD) {
            __builtin_prefetch(hs + kk + 32 + koffA, 0, 3);
            __builtin_prefetch(hl + kk + 32 + koffA, 0, 3);
        }
        v16bf As, Al;
        {
            v8bf c1 = *(const v8bf*)(hs + kk + koffA);
            v8bf c2 = *(const v8bf*)(hs + kk + 16 + koffA);
            v8bf d1 = *(const v8bf*)(hl + kk + koffA);
            v8bf d2 = *(const v8bf*)(hl + kk + 16 + koffA);
#pragma unroll
            for (int i = 0; i < 8; i++) {
                As[i] = c1[i]; As[8 + i] = c2[i];
                Al[i] = d1[i]; Al[8 + i] = d2[i];
            }
        }

        auto loadB2 = [&](int sn, v16bf& Bs, v16bf& Bl) {
            const char* bpS = sB + cur * (2 * TILEB) + (sn * 16 + lr) * BROW + koffB2;
            const char* bpL = bpS + TILEB;
            v8bf s1 = *(const v8bf*)bpS;
            v8bf s2 = *(const v8bf*)(bpS + 32);
            v8bf l1 = *(const v8bf*)bpL;
            v8bf l2 = *(const v8bf*)(bpL + 32);
#pragma unroll
            for (int i = 0; i < 8; i++) {
                Bs[i] = s1[i]; Bs[8 + i] = s2[i];
                Bl[i] = l1[i]; Bl[8 + i] = l2[i];
            }
        };

        v16bf BsC, BlC;
        loadB2(0, BsC, BlC);
#pragma unroll
        for (int sn = 0; sn < 8; sn++) {
            v16bf BsN, BlN;
            if (sn < 7) loadB2(sn + 1, BsN, BlN);
            accS[sn] = __builtin_amdgcn_wmma_f32_16x16x32_bf16(
                false, As, false, BsC, (short)0, accS[sn], false, false);
            accL[sn] = __builtin_amdgcn_wmma_f32_16x16x32_bf16(
                false, Al, false, BlC, (short)0, accL[sn], false, false);
            BsC = BsN; BlC = BlN;
        }
    }

    // Epilogue: integrated = r0*(yS+bS) + r1*(yL+bL) + r2*(formal@ffw^T + ffb)
#pragma unroll
    for (int j = 0; j < 8; j++) {
        int t = t0 + half * 8 + j;
        float r0 = rpl[t], r1 = rpl[TT + t], r2 = rpl[2 * TT + t];
        float f0 = fpl[t], f1 = fpl[TT + t], f2 = fpl[2 * TT + t];
#pragma unroll
        for (int sn = 0; sn < 8; sn++) {
            int o = n0 + sn * 16 + lr;
            float oform = f0 * ffw[o * 3] + f1 * ffw[o * 3 + 1] + f2 * ffw[o * 3 + 2] + ffb[o];
            float v = r0 * (accS[sn][j] + syn_out_b[o])
                    + r1 * (accL[sn][j] + log_out_b[o])
                    + r2 * oform;
            out[(size_t)t * DD + o] = v;
        }
    }
}

// ---------------------------------------------------------------------------
extern "C" void kernel_launch(void* const* d_in, const int* in_sizes, int n_in,
                              void* d_out, int out_size, void* d_ws, size_t ws_size,
                              hipStream_t stream) {
    (void)in_sizes; (void)n_in; (void)out_size; (void)ws_size;
    const float* x         = (const float*)d_in[0];
    const float* gate_w    = (const float*)d_in[1];
    const float* gate_b    = (const float*)d_in[2];
    const float* syn_ll    = (const float*)d_in[3];
    const float* syn_in_w  = (const float*)d_in[4];
    const float* syn_in_b  = (const float*)d_in[5];
    const float* syn_out_w = (const float*)d_in[6];
    const float* syn_out_b = (const float*)d_in[7];
    const float* log_ll    = (const float*)d_in[8];
    const float* log_in_w  = (const float*)d_in[9];
    const float* log_in_b  = (const float*)d_in[10];
    const float* log_out_w = (const float*)d_in[11];
    const float* log_out_b = (const float*)d_in[12];
    const float* to_w      = (const float*)d_in[13];
    const float* to_b      = (const float*)d_in[14];
    const float* ffw       = (const float*)d_in[15];
    const float* ffb       = (const float*)d_in[16];
    float* out = (float*)d_out;

    char* ws = (char*)d_ws;
    __bf16* Win   = (__bf16*)(ws);                                   // 1280*640 bf16
    __bf16* Wout  = (__bf16*)(ws + (size_t)NCAT * DD * 2);           // 1280*640 bf16
    __bf16* u_syn = (__bf16*)(ws + (size_t)2 * NCAT * DD * 2);       // T*640 bf16
    __bf16* u_log = u_syn + (size_t)TT * DD;                         // T*640 bf16
    float*  rpl   = (float*)(u_log + (size_t)TT * DD);               // 3*T f32
    float*  fpl   = rpl + (size_t)3 * TT;                            // 3*T f32

    k_cvt_weights<<<(2 * NCAT * DD) / 256, 256, 0, stream>>>(
        syn_in_w, log_in_w, syn_out_w, log_out_w, Win, Wout);

    k_gates<<<TT / 8, 256, 0, stream>>>(x, gate_w, gate_b, to_w, to_b, rpl, fpl);

    k_in_gemm<<<dim3(TT / 128, NCAT / 128), 256, 0, stream>>>(
        x, Win, syn_in_b, log_in_b, u_syn, u_log);

    k_scan<<<(2 * NB * 320) / 256, 256, 0, stream>>>(syn_ll, log_ll, u_syn, u_log, out);

    k_out_gemm<<<dim3(TT / 128, DD / 128), 256, 0, stream>>>(
        u_syn, u_log, Wout, syn_out_b, log_out_b, ffw, ffb, rpl, fpl, out);
}
